// MarkdownModel_73581379715314
// MI455X (gfx1250) — matrix operand order repaired and checked by reading, not compiled, not run on hardware
//
#include <hip/hip_runtime.h>
#include <cstdint>
#include <cstddef>

// ---------------- types ----------------
typedef __attribute__((ext_vector_type(16))) __bf16        v16bf;
typedef __attribute__((ext_vector_type(8)))  float         v8f;
typedef __attribute__((ext_vector_type(4)))  unsigned int  u32x4;
typedef __attribute__((ext_vector_type(8)))  unsigned int  u32x8;
typedef __attribute__((ext_vector_type(4)))  int           i32x4;
typedef __attribute__((ext_vector_type(8)))  int           i32x8;

static constexpr int B_  = 16;
static constexpr int S_  = 2048;
static constexpr int D_  = 768;
static constexpr int NC_ = 32;
static constexpr int NM_ = 16;
static constexpr int KP_ = 800;    // padded K for the top projection (>=772, mult of 32)
static constexpr int G3_ = 3 * D_; // 2304 gate width
static constexpr int NB_ = B_ * NM_; // 256 GRU batch rows
static constexpr int T_  = NC_;      // 32 time steps
static constexpr int WLDS_ = 6 * 16 * D_; // staged weight elements per block (144 KB bf16)

// ---------------- helpers ----------------
__device__ __forceinline__ unsigned short f2bf(float f) {
  union { float f; unsigned int u; } v; v.f = f;
  unsigned int u = v.u;
  unsigned int r = u + 0x7fffu + ((u >> 16) & 1u);
  if ((u & 0x7fffffffu) > 0x7f800000u) r = u | 0x00400000u; // quiet NaN
  return (unsigned short)(r >> 16);
}

__device__ __forceinline__ float sigmoidf_(float x) {
  return 1.0f / (1.0f + __expf(-x));
}

// A-matrix 16x32 bf16 tile (ISA 7.12.2): lanes0-15: M=lane, K 0..7 & 16..23;
// lanes16-31: M=lane-16, K 8..15 & 24..31.
__device__ __forceinline__ v16bf load_a_tile_bf(const unsigned short* base, int stride) {
  int lane = threadIdx.x & 31;
  const unsigned short* p = base + (size_t)(lane & 15) * stride + ((lane >> 4) << 3);
  u32x4 lo = *(const u32x4*)p;
  u32x4 hi = *(const u32x4*)(p + 16);
  u32x8 u = {lo[0], lo[1], lo[2], lo[3], hi[0], hi[1], hi[2], hi[3]};
  return __builtin_bit_cast(v16bf, u);
}

// Same tile but source is f32 (recurrent h); convert to bf16 in-register.
__device__ __forceinline__ v16bf load_a_tile_f32(const float* base, int stride) {
  int lane = threadIdx.x & 31;
  const float* p = base + (size_t)(lane & 15) * stride + ((lane >> 4) << 3);
  unsigned short t[16];
#pragma unroll
  for (int i = 0; i < 8; ++i) t[i]     = f2bf(p[i]);
#pragma unroll
  for (int i = 0; i < 8; ++i) t[8 + i] = f2bf(p[16 + i]);
  u32x8 u;
#pragma unroll
  for (int i = 0; i < 8; ++i) u[i] = (unsigned int)t[2 * i] | ((unsigned int)t[2 * i + 1] << 16);
  return __builtin_bit_cast(v16bf, u);
}

// B-matrix 32x16 bf16 tile: B[k,n] = W[n0+n, k0+k]; lanes0-15: n=lane, K 0..7 & 16..23;
// lanes16-31: n=lane-16, K 8..15 & 24..31.
__device__ __forceinline__ v16bf load_b_tile_bf(const unsigned short* base, int stride) {
  int lane = threadIdx.x & 31;
  const unsigned short* p = base + (size_t)(lane & 15) * stride + ((lane >> 4) << 3);
  u32x4 lo = *(const u32x4*)p;
  u32x4 hi = *(const u32x4*)(p + 16);
  u32x8 u = {lo[0], lo[1], lo[2], lo[3], hi[0], hi[1], hi[2], hi[3]};
  return __builtin_bit_cast(v16bf, u);
}

// Same B tile, but sourced from LDS-staged weights (16 rows x 768 cols, row-major).
__device__ __forceinline__ v16bf load_b_tile_lds(const unsigned short* base, int k) {
  int lane = threadIdx.x & 31;
  const unsigned short* p = base + (lane & 15) * D_ + k + ((lane >> 4) << 3);
  u32x4 lo = *(const u32x4*)p;
  u32x4 hi = *(const u32x4*)(p + 16);
  u32x8 u = {lo[0], lo[1], lo[2], lo[3], hi[0], hi[1], hi[2], hi[3]};
  return __builtin_bit_cast(v16bf, u);
}

__device__ __forceinline__ v8f wmma_bf16(v16bf a, v16bf b, v8f c) {
  return __builtin_amdgcn_wmma_f32_16x16x32_bf16(false, a, false, b, (short)0, c, false, false);
}

#if __has_builtin(__builtin_amdgcn_tensor_load_to_lds)
// TDM 2D tile load: 16 rows x 768 bf16 cols, row stride 768, global -> LDS.
// D# built per CDNA5 ISA 8.3/8.4 (group0: count|lds_addr|global_addr|type=2,
// group1: data_size=2B, tensor_dim0=768, tensor_dim1=2304, tile 768x16,
// tensor_dim0_stride=768). Groups 2/3 unused (2D tensor).
// This toolchain exposes the 6-arg builtin:
//   (uint32x4 g0, int32x8 g1, int32x4 g2, int32x4 g3, int32x8 g4, i32 cpol)
struct TdmG0 { unsigned long long q0, q1; };
struct TdmG1 { unsigned long long q0, q1, q2, q3; };
__device__ __forceinline__ void tdm_load_tile16(unsigned int lds_off, const void* gsrc) {
  unsigned long long ga = (unsigned long long)(uintptr_t)gsrc;
  TdmG0 G0;
  G0.q0 = 1ull | ((unsigned long long)lds_off << 32);                 // count=1, lds_addr
  G0.q1 = (ga & 0x01FFFFFFFFFFFFFFull) | (2ull << 62);                // global_addr, type=2
  TdmG1 G1;
  G1.q0 = (1ull << 16) | (768ull << 48);        // data_size=2B; tensor_dim0[15:0]
  G1.q1 = (2304ull << 16) | (768ull << 48);     // tensor_dim0 hi=0; tensor_dim1; tile_dim0
  G1.q2 = 16ull | (768ull << 32);               // tile_dim1=16; tensor_dim0_stride lo32
  G1.q3 = 0ull;
  u32x4 g0 = __builtin_bit_cast(u32x4, G0);
  i32x8 g1 = __builtin_bit_cast(i32x8, G1);
  i32x4 z4 = {0, 0, 0, 0};
  i32x8 z8 = {0, 0, 0, 0, 0, 0, 0, 0};
  __builtin_amdgcn_tensor_load_to_lds(g0, g1, z4, z4, z8, 0);
}
#endif

// ---------------- kernel 1: segment means -> bf16 feature rows ----------------
__global__ void seg_feat_kernel(const float* __restrict__ x, const int* __restrict__ gids,
                                const float* __restrict__ fts, const float* __restrict__ cpr,
                                unsigned short* __restrict__ code_feat,
                                unsigned short* __restrict__ md_feat) {
  int b = blockIdx.x / 48;
  int seg = blockIdx.x % 48;
  int target = (seg < 32) ? -(seg + 1) : (seg - 31);
  const float* xb = x + (size_t)b * S_ * D_;
  const int* gb = gids + b * S_;
  float a0 = 0.f, a1 = 0.f, a2 = 0.f;
  int cnt = 0;
  int tid = threadIdx.x;
  for (int s = 0; s < S_; ++s) {
    if (gb[s] == target) {
      ++cnt;
      const float* row = xb + (size_t)s * D_;
      a0 += row[tid];
      a1 += row[tid + 256];
      a2 += row[tid + 512];
    }
  }
  float inv = 1.0f / (float)cnt;
  unsigned short* out = (seg < 32) ? (code_feat + (size_t)(b * NC_ + seg) * KP_)
                                   : (md_feat + (size_t)(b * NM_ + (seg - 32)) * KP_);
  out[tid]       = f2bf(a0 * inv);
  out[tid + 256] = f2bf(a1 * inv);
  out[tid + 512] = f2bf(a2 * inv);
  if (tid < 32) {
    float v = 0.f;
    if (seg < 32) { if (tid == 0) v = cpr[b * NC_ + seg]; }
    else          { if (tid < 4)  v = fts[b * 4 + tid]; }
    out[D_ + tid] = f2bf(v);
  }
}

// ---------------- kernel 2: generic f32 -> padded bf16 pack ----------------
__global__ void pack_bf16_kernel(const float* __restrict__ src, unsigned short* __restrict__ dst,
                                 int rows, int src_stride, int col0, int ncols, int dst_stride) {
  long long idx = (long long)blockIdx.x * blockDim.x + threadIdx.x;
  long long total = (long long)rows * dst_stride;
  if (idx >= total) return;
  int r = (int)(idx / dst_stride);
  int c = (int)(idx % dst_stride);
  dst[idx] = (c < ncols) ? f2bf(src[(size_t)r * src_stride + col0 + c]) : (unsigned short)0;
}

// ---------------- kernel 3: WMMA GEMM C[M,N] = A[M,K] * W[N,K]^T (+bias) -----
__global__ void gemm_bf16_kernel(const unsigned short* __restrict__ A,
                                 const unsigned short* __restrict__ W,
                                 const float* __restrict__ bias, float* __restrict__ C,
                                 int M, int N, int K) {
  int wave = (int)((blockIdx.x * blockDim.x + threadIdx.x) >> 5);
  int npairs = N >> 5;
  int mt = wave / npairs, np = wave % npairs;
  int m0 = mt << 4, n0 = np << 5;
  if (m0 >= M) return;
  v8f acc0 = {}, acc1 = {};
  for (int k = 0; k < K; k += 32) {
    v16bf a  = load_a_tile_bf(A + (size_t)m0 * K + k, K);
    v16bf b0 = load_b_tile_bf(W + (size_t)n0 * K + k, K);
    v16bf b1 = load_b_tile_bf(W + (size_t)(n0 + 16) * K + k, K);
    acc0 = wmma_bf16(a, b0, acc0);
    acc1 = wmma_bf16(a, b1, acc1);
  }
  int lane = threadIdx.x & 31;
  int n = lane & 15, mo = (lane >> 4) << 3;
  float bv0 = bias ? bias[n0 + n] : 0.f;
  float bv1 = bias ? bias[n0 + 16 + n] : 0.f;
#pragma unroll
  for (int r = 0; r < 8; ++r) {
    int m = m0 + r + mo;
    C[(size_t)m * N + n0 + n]      = acc0[r] + bv0;
    C[(size_t)m * N + n0 + 16 + n] = acc1[r] + bv1;
  }
}

// ---------------- kernel 4: merged = relu(Pmd + Pcode), time-major bf16 ------
__global__ void merged_kernel(const float* __restrict__ Pmd, const float* __restrict__ Pcode,
                              unsigned short* __restrict__ Xbf) {
  int idx = blockIdx.x;               // b*512 + m*32 + c
  int b = idx >> 9, m = (idx >> 5) & 15, c = idx & 31;
  const float* pm = Pmd + (size_t)(b * NM_ + m) * D_;
  const float* pc = Pcode + (size_t)(b * NC_ + c) * D_;
  unsigned short* out = Xbf + ((size_t)c * NB_ + (b * NM_ + m)) * D_;
  for (int d = threadIdx.x; d < D_; d += blockDim.x) {
    float v = pm[d] + pc[d];
    out[d] = f2bf(v > 0.f ? v : 0.f);
  }
}

// ---------------- kernel 5: one GRU step -------------------------------------
// Block = 512 threads (16 waves) = one h-tile (16 hidden units) x all 256 rows.
// The 6 gate weight tiles (w_ih/w_hh r,z,n rows j0..j0+15, full K=768) are
// staged in LDS once per block via the Tensor Data Mover, then shared by all
// 16 waves; each wave runs 6 WMMA accumulator chains over K.
__global__ void __launch_bounds__(512) gru_step_kernel(
    const unsigned short* __restrict__ Xt,   // [256,768] bf16, time slice
    const float* __restrict__ h_prev,        // [256,768] f32 or nullptr
    const unsigned short* __restrict__ w_ih, // [2304,768] bf16
    const unsigned short* __restrict__ w_hh,
    const float* __restrict__ b_ih, const float* __restrict__ b_hh,
    float* __restrict__ h_out) {             // [256,768]
  extern __shared__ unsigned short wsm[];    // [6][16][768] bf16 = 144 KB
  int ht = blockIdx.x;                       // 0..47
  int j0 = ht << 4;
  int waveId = (int)(threadIdx.x >> 5);      // 0..15

#if __has_builtin(__builtin_amdgcn_tensor_load_to_lds)
  if (waveId == 0) {
#pragma unroll
    for (int g = 0; g < 6; ++g) {
      const unsigned short* src = (g < 3 ? w_ih : w_hh) + (size_t)((g % 3) * D_ + j0) * D_;
      tdm_load_tile16((unsigned int)(uintptr_t)&wsm[g * 16 * D_], (const void*)src);
    }
    __builtin_amdgcn_s_wait_tensorcnt(0);
  }
#else
  for (int i = threadIdx.x; i < WLDS_; i += 512) {
    int g = i / (16 * D_);
    int rr = (i / D_) % 16;
    int c = i % D_;
    wsm[i] = (g < 3 ? w_ih : w_hh)[(size_t)((g % 3) * D_ + j0 + rr) * D_ + c];
  }
#endif
  __syncthreads();

  int m0 = waveId << 4;
  v8f axr = {}, axz = {}, axn = {}, ahr = {}, ahz = {}, ahn = {};
  for (int k = 0; k < D_; k += 32) {
    v16bf ax = load_a_tile_bf(Xt + (size_t)m0 * D_ + k, D_);
    v16bf br = load_b_tile_lds(wsm + 0 * 16 * D_, k);
    v16bf bz = load_b_tile_lds(wsm + 1 * 16 * D_, k);
    v16bf bn = load_b_tile_lds(wsm + 2 * 16 * D_, k);
    axr = wmma_bf16(ax, br, axr);
    axz = wmma_bf16(ax, bz, axz);
    axn = wmma_bf16(ax, bn, axn);
    if (h_prev) {
      v16bf ah = load_a_tile_f32(h_prev + (size_t)m0 * D_ + k, D_);
      v16bf cr = load_b_tile_lds(wsm + 3 * 16 * D_, k);
      v16bf cz = load_b_tile_lds(wsm + 4 * 16 * D_, k);
      v16bf cn = load_b_tile_lds(wsm + 5 * 16 * D_, k);
      ahr = wmma_bf16(ah, cr, ahr);
      ahz = wmma_bf16(ah, cz, ahz);
      ahn = wmma_bf16(ah, cn, ahn);
    }
  }
  int lane = threadIdx.x & 31;
  int n = lane & 15, mo = (lane >> 4) << 3;
  int j = j0 + n;
  float bir = b_ih[j], biz = b_ih[D_ + j], bin_ = b_ih[2 * D_ + j];
  float bhr = b_hh[j], bhz = b_hh[D_ + j], bhn = b_hh[2 * D_ + j];
#pragma unroll
  for (int r = 0; r < 8; ++r) {
    int m = m0 + r + mo;
    float rg = sigmoidf_((axr[r] + bir) + (ahr[r] + bhr));
    float zg = sigmoidf_((axz[r] + biz) + (ahz[r] + bhz));
    float ng = tanhf((axn[r] + bin_) + rg * (ahn[r] + bhn));
    float hp = h_prev ? h_prev[(size_t)m * D_ + j] : 0.f;
    h_out[(size_t)m * D_ + j] = (1.f - zg) * ng + zg * hp;
  }
}

// ---------------- kernel 6: final projection with wave reduction -------------
__global__ void preds_kernel(const float* __restrict__ ysf, const float* __restrict__ ysb,
                             const float* __restrict__ w2, const float* __restrict__ b2,
                             float* __restrict__ out) {
  int wave = (int)((blockIdx.x * blockDim.x + threadIdx.x) >> 5); // 0..8191
  int lane = threadIdx.x & 31;
  int b = wave >> 9, m = (wave >> 5) & 15, c = wave & 31;
  size_t row = (size_t)c * NB_ + (b * NM_ + m);
  const float* f = ysf + row * D_;
  const float* g = ysb + row * D_;
  float s = 0.f;
  for (int j = lane; j < D_; j += 32) s += f[j] * w2[j] + g[j] * w2[D_ + j];
#pragma unroll
  for (int o = 16; o > 0; o >>= 1) s += __shfl_down(s, o, 32);
  if (lane == 0) out[(size_t)(b * NM_ + m) * NC_ + c] = s + b2[0];
}

// ---------------- launch ----------------
extern "C" void kernel_launch(void* const* d_in, const int* in_sizes, int n_in,
                              void* d_out, int out_size, void* d_ws, size_t ws_size,
                              hipStream_t stream) {
  const float* x     = (const float*)d_in[0];
  const float* fts   = (const float*)d_in[1];
  const float* cpr   = (const float*)d_in[2];
  const int*   gids  = (const int*)d_in[3];
  const float* w_top = (const float*)d_in[4];
  const float* b_top = (const float*)d_in[5];
  const float* w_ih_f = (const float*)d_in[6];
  const float* w_hh_f = (const float*)d_in[7];
  const float* b_ih_f = (const float*)d_in[8];
  const float* b_hh_f = (const float*)d_in[9];
  const float* w_ih_b = (const float*)d_in[10];
  const float* w_hh_b = (const float*)d_in[11];
  const float* b_ih_b = (const float*)d_in[12];
  const float* b_hh_b = (const float*)d_in[13];
  const float* w_top2 = (const float*)d_in[14];
  const float* b_top2 = (const float*)d_in[15];
  float* out = (float*)d_out;

  char* ws = (char*)d_ws;
  size_t off = 0;
  auto take = [&](size_t bytes) { char* p = ws + off; off = (off + bytes + 255) & ~(size_t)255; return p; };
  unsigned short* code_feat = (unsigned short*)take((size_t)B_ * NC_ * KP_ * 2);
  unsigned short* md_feat   = (unsigned short*)take((size_t)B_ * NM_ * KP_ * 2);
  unsigned short* Wm_bf     = (unsigned short*)take((size_t)D_ * KP_ * 2);
  unsigned short* Wc_bf     = (unsigned short*)take((size_t)D_ * KP_ * 2);
  float*          Pmd       = (float*)take((size_t)B_ * NM_ * D_ * 4);
  float*          Pcode     = (float*)take((size_t)B_ * NC_ * D_ * 4);
  unsigned short* Xbf       = (unsigned short*)take((size_t)T_ * NB_ * D_ * 2);
  unsigned short* wihf_bf   = (unsigned short*)take((size_t)G3_ * D_ * 2);
  unsigned short* whhf_bf   = (unsigned short*)take((size_t)G3_ * D_ * 2);
  unsigned short* wihb_bf   = (unsigned short*)take((size_t)G3_ * D_ * 2);
  unsigned short* whhb_bf   = (unsigned short*)take((size_t)G3_ * D_ * 2);
  float*          ysf       = (float*)take((size_t)T_ * NB_ * D_ * 4);
  float*          ysb       = (float*)take((size_t)T_ * NB_ * D_ * 4);
  (void)ws_size; (void)in_sizes; (void)n_in; (void)out_size;

  // 1) segment means + feature assembly
  seg_feat_kernel<<<B_ * 48, 256, 0, stream>>>(x, gids, fts, cpr, code_feat, md_feat);

  // 2) weight packing to bf16
  auto pack = [&](const float* src, unsigned short* dst, int rows, int ss, int c0, int nc, int ds) {
    long long total = (long long)rows * ds;
    int blocks = (int)((total + 255) / 256);
    pack_bf16_kernel<<<blocks, 256, 0, stream>>>(src, dst, rows, ss, c0, nc, ds);
  };
  pack(w_top, Wm_bf, D_, 2 * D_ + 5, 0, 772, KP_);     // md part of w_top
  pack(w_top, Wc_bf, D_, 2 * D_ + 5, 772, 769, KP_);   // code part of w_top
  pack(w_ih_f, wihf_bf, G3_, D_, 0, D_, D_);
  pack(w_hh_f, whhf_bf, G3_, D_, 0, D_, D_);
  pack(w_ih_b, wihb_bf, G3_, D_, 0, D_, D_);
  pack(w_hh_b, whhb_bf, G3_, D_, 0, D_, D_);

  // 3) top projections (bias folded into the md part)
  {
    int waves = (B_ * NM_ / 16) * (D_ / 32);   // 384
    gemm_bf16_kernel<<<waves * 32 / 256, 256, 0, stream>>>(md_feat, Wm_bf, b_top, Pmd,
                                                           B_ * NM_, D_, KP_);
    waves = (B_ * NC_ / 16) * (D_ / 32);       // 768
    gemm_bf16_kernel<<<waves * 32 / 256, 256, 0, stream>>>(code_feat, Wc_bf, nullptr, Pcode,
                                                           B_ * NC_, D_, KP_);
  }

  // 4) merged = relu(Pmd + Pcode) -> time-major bf16
  merged_kernel<<<B_ * NM_ * NC_, 256, 0, stream>>>(Pmd, Pcode, Xbf);

  // 5) GRU, forward then backward: 48 blocks x 512 threads, 144 KB dynamic LDS
  const size_t lds_bytes = (size_t)WLDS_ * 2;
  for (int t = 0; t < T_; ++t) {
    const float* hp = (t == 0) ? nullptr : (ysf + (size_t)(t - 1) * NB_ * D_);
    gru_step_kernel<<<48, 512, lds_bytes, stream>>>(Xbf + (size_t)t * NB_ * D_, hp,
                                                    wihf_bf, whhf_bf, b_ih_f, b_hh_f,
                                                    ysf + (size_t)t * NB_ * D_);
  }
  for (int tt = 0; tt < T_; ++tt) {
    int t = T_ - 1 - tt;
    const float* hp = (tt == 0) ? nullptr : (ysb + (size_t)(t + 1) * NB_ * D_);
    gru_step_kernel<<<48, 512, lds_bytes, stream>>>(Xbf + (size_t)t * NB_ * D_, hp,
                                                    wihb_bf, whhb_bf, b_ih_b, b_hh_b,
                                                    ysb + (size_t)t * NB_ * D_);
  }

  // 6) final projection
  preds_kernel<<<(B_ * NM_ * NC_ * 32) / 256, 256, 0, stream>>>(ysf, ysb, w_top2, b_top2, out);
}
